// BidirectionalAttention_54425825575691
// MI455X (gfx1250) — matrix operand order repaired and checked by reading, not compiled, not run on hardware
//
#include <hip/hip_runtime.h>

#define B_ 32
#define C_ 1024
#define Q_ 128
#define E_ 512

typedef __attribute__((ext_vector_type(16))) __bf16 bf16x16;
typedef __attribute__((ext_vector_type(8)))  float  f32x8;
typedef __attribute__((ext_vector_type(4)))  unsigned int u32x4;

union FragU { u32x4 q[2]; bf16x16 v; };

__device__ __forceinline__ unsigned short f2bf(float x) {
  unsigned u = __float_as_uint(x);
  u += 0x7FFFu + ((u >> 16) & 1u);          // round-to-nearest-even
  return (unsigned short)(u >> 16);
}
__device__ __forceinline__ unsigned packbf(float a, float b) {
  return (unsigned)f2bf(a) | ((unsigned)f2bf(b) << 16);
}

// ---------------------------------------------------------------------------
// Kernel 1: sim = qdot + cdot + (ctx .* ws_m) @ qu^T ; softmax_q ; c2q = P@qu
// One block = one batch b x 128 context rows. 256 threads = 8 waves (wave32).
// ---------------------------------------------------------------------------
__global__ __launch_bounds__(256) void bidaf_sim_c2q_kernel(
    const float* __restrict__ ctx, const float* __restrict__ qu,
    const float* __restrict__ ws, float* __restrict__ out,
    float* __restrict__ rowmax_ws) {
  __shared__ unsigned A_lds[128 * 20];        // 128 rows x 32 bf16, 80B stride
  __shared__ unsigned Bq_lds[128 * 20];       // question chunk / QT chunk
  __shared__ unsigned short P_lds[128 * 136]; // 128x128 bf16, 272B stride
  __shared__ float wsm_lds[E_];
  __shared__ float qdot_lds[128];
  __shared__ float cdot_lds[128];
  __shared__ float red[256];

  const int tid  = threadIdx.x;
  const int lane = tid & 31;
  const int w    = tid >> 5;       // wave 0..7 -> owns c rows 16w..16w+15
  const int lh   = lane & 15;      // lane within half
  const int half = lane >> 4;
  const int cblk = blockIdx.x;     // 0..7
  const int b    = blockIdx.y;     // 0..31
  const int cbase = cblk * 128;

  const float* wsq  = ws;
  const float* wsc  = ws + E_;
  const float* wsmg = ws + 2 * E_;

  for (int i = tid; i < E_; i += 256) wsm_lds[i] = wsmg[i];

  // ---- qdot[q] = qu[q,:].ws_q ; cdot[c] = ctx[c,:].ws_c (2 threads/row) ----
  {
    const int r = tid >> 1, h = tid & 1;
    const float4* qp = (const float4*)(qu + ((size_t)(b * Q_) + r) * E_ + h * 256);
    const float4* wp = (const float4*)(wsq + h * 256);
    float s = 0.f;
    #pragma unroll 4
    for (int i = 0; i < 64; ++i) {
      float4 a = qp[i], c = wp[i];
      s += a.x * c.x + a.y * c.y + a.z * c.z + a.w * c.w;
    }
    red[tid] = s;
  }
  __syncthreads();
  if (tid < 128) qdot_lds[tid] = red[2 * tid] + red[2 * tid + 1];
  __syncthreads();
  {
    const int r = tid >> 1, h = tid & 1;
    const float4* cp = (const float4*)(ctx + ((size_t)(b * C_) + cbase + r) * E_ + h * 256);
    const float4* wp = (const float4*)(wsc + h * 256);
    float s = 0.f;
    #pragma unroll 4
    for (int i = 0; i < 64; ++i) {
      float4 a = cp[i], c = wp[i];
      s += a.x * c.x + a.y * c.y + a.z * c.z + a.w * c.w;
    }
    red[tid] = s;
  }
  __syncthreads();
  if (tid < 128) cdot_lds[tid] = red[2 * tid] + red[2 * tid + 1];

  // ---- GEMM1: S(128x128) = bf16(ctx .* ws_m) @ bf16(qu)^T, K = 512 ----
  f32x8 acc[8];
  #pragma unroll
  for (int j = 0; j < 8; ++j) acc[j] = (f32x8)(0.0f);

  for (int kk = 0; kk < E_; kk += 32) {
    __syncthreads();
    #pragma unroll
    for (int i = 0; i < 8; ++i) {
      int idx = tid + 256 * i;             // 2048 float2 elements
      int row = idx >> 4;
      int c2  = idx & 15;
      int e   = kk + (c2 << 1);
      float2 cv = *(const float2*)(ctx + ((size_t)(b * C_) + cbase + row) * E_ + e);
      A_lds[row * 20 + c2]  = packbf(cv.x * wsm_lds[e], cv.y * wsm_lds[e + 1]);
      float2 qv = *(const float2*)(qu + ((size_t)(b * Q_) + row) * E_ + e);
      Bq_lds[row * 20 + c2] = packbf(qv.x, qv.y);
    }
    __syncthreads();
    FragU fa;
    {
      const char* base = (const char*)A_lds + (16 * w + lh) * 80 + (half << 4);
      fa.q[0] = *(const u32x4*)(base);
      fa.q[1] = *(const u32x4*)(base + 32);
    }
    #pragma unroll
    for (int j = 0; j < 8; ++j) {
      FragU fb;
      const char* base = (const char*)Bq_lds + (16 * j + lh) * 80 + (half << 4);
      fb.q[0] = *(const u32x4*)(base);
      fb.q[1] = *(const u32x4*)(base + 32);
      acc[j] = __builtin_amdgcn_wmma_f32_16x16x32_bf16(
          false, fa.v, false, fb.v, (short)0, acc[j], false, false);
    }
  }

  // ---- bias add + softmax over q (rows live in 16-lane halves) ----
  float cdl[8];
  #pragma unroll
  for (int r = 0; r < 8; ++r) cdl[r] = cdot_lds[16 * w + 8 * half + r];
  #pragma unroll
  for (int j = 0; j < 8; ++j) {
    float qd = qdot_lds[16 * j + lh];
    #pragma unroll
    for (int r = 0; r < 8; ++r) acc[j][r] += qd + cdl[r];
  }
  float mr[8], sr[8];
  #pragma unroll
  for (int r = 0; r < 8; ++r) {
    float m = acc[0][r];
    #pragma unroll
    for (int j = 1; j < 8; ++j) m = fmaxf(m, acc[j][r]);
    m = fmaxf(m, __shfl_xor(m, 1, 32));
    m = fmaxf(m, __shfl_xor(m, 2, 32));
    m = fmaxf(m, __shfl_xor(m, 4, 32));
    m = fmaxf(m, __shfl_xor(m, 8, 32));
    mr[r] = m;
  }
  #pragma unroll
  for (int r = 0; r < 8; ++r) {
    float s = 0.f;
    #pragma unroll
    for (int j = 0; j < 8; ++j) {
      float p = __expf(acc[j][r] - mr[r]);
      acc[j][r] = p;
      s += p;
    }
    s += __shfl_xor(s, 1, 32);
    s += __shfl_xor(s, 2, 32);
    s += __shfl_xor(s, 4, 32);
    s += __shfl_xor(s, 8, 32);
    sr[r] = 1.0f / s;
  }
  #pragma unroll
  for (int r = 0; r < 8; ++r) {
    int row = 16 * w + 8 * half + r;
    #pragma unroll
    for (int j = 0; j < 8; ++j)
      P_lds[row * 136 + 16 * j + lh] = f2bf(acc[j][r] * sr[r]);
  }
  if (lh == 0) {
    #pragma unroll
    for (int r = 0; r < 8; ++r)
      rowmax_ws[(size_t)b * C_ + cbase + 16 * w + 8 * half + r] = mr[r];
  }

  // ---- GEMM2: c2q(128x512) = P(128x128,bf16) @ qu(128x512), K = 128 ----
  for (int e0 = 0; e0 < E_; e0 += 128) {
    f32x8 acc2[8];
    #pragma unroll
    for (int j = 0; j < 8; ++j) acc2[j] = (f32x8)(0.0f);
    for (int kq = 0; kq < 128; kq += 32) {
      __syncthreads();
      unsigned short* Bh = (unsigned short*)Bq_lds;  // QT: [e'][q'] bf16
      #pragma unroll
      for (int i = 0; i < 8; ++i) {
        int idx  = tid + 256 * i;       // 2048 float2 elements
        int qrow = idx >> 6;            // 0..31
        int e2   = idx & 63;
        int e    = e0 + (e2 << 1);
        float2 qv = *(const float2*)(qu + ((size_t)(b * Q_) + kq + qrow) * E_ + e);
        Bh[(e2 * 2)     * 40 + qrow] = f2bf(qv.x);
        Bh[(e2 * 2 + 1) * 40 + qrow] = f2bf(qv.y);
      }
      __syncthreads();
      FragU fa;
      {
        const char* base = (const char*)P_lds + (16 * w + lh) * 272 + (kq << 1) + (half << 4);
        fa.q[0] = *(const u32x4*)(base);
        fa.q[1] = *(const u32x4*)(base + 32);
      }
      #pragma unroll
      for (int j = 0; j < 8; ++j) {
        FragU fb;
        const char* base = (const char*)Bq_lds + (16 * j + lh) * 80 + (half << 4);
        fb.q[0] = *(const u32x4*)(base);
        fb.q[1] = *(const u32x4*)(base + 32);
        acc2[j] = __builtin_amdgcn_wmma_f32_16x16x32_bf16(
            false, fa.v, false, fb.v, (short)0, acc2[j], false, false);
      }
    }
    #pragma unroll
    for (int j = 0; j < 8; ++j) {
      int e = e0 + 16 * j + lh;
      #pragma unroll
      for (int r = 0; r < 8; ++r) {
        int c = cbase + 16 * w + 8 * half + r;
        out[((size_t)(b * C_) + c) * E_ + e] = acc2[j][r];
      }
    }
  }
}

// ---------------------------------------------------------------------------
// Kernel 2: q2c_w = softmax_c(rowmax) ; q2c[b,e] = sum_c w[c]*ctx[b,c,e]
// ---------------------------------------------------------------------------
__global__ __launch_bounds__(256) void bidaf_q2c_kernel(
    const float* __restrict__ ctx, const float* __restrict__ rowmax_ws,
    float* __restrict__ out) {
  __shared__ float wgt[C_];
  __shared__ float red[256];
  const int tid = threadIdx.x;
  const int b   = blockIdx.x;
  const float* rm = rowmax_ws + (size_t)b * C_;

  float lm = -3.4e38f;
  for (int c = tid; c < C_; c += 256) lm = fmaxf(lm, rm[c]);
  red[tid] = lm;
  __syncthreads();
  for (int s = 128; s > 0; s >>= 1) {
    if (tid < s) red[tid] = fmaxf(red[tid], red[tid + s]);
    __syncthreads();
  }
  float M = red[0];
  __syncthreads();

  float ls = 0.f;
  for (int c = tid; c < C_; c += 256) {
    float e = __expf(rm[c] - M);
    wgt[c] = e;
    ls += e;
  }
  red[tid] = ls;
  __syncthreads();
  for (int s = 128; s > 0; s >>= 1) {
    if (tid < s) red[tid] += red[tid + s];
    __syncthreads();
  }
  float inv = 1.0f / red[0];

  for (int e = tid; e < E_; e += 256) {
    const float* cp = ctx + (size_t)b * C_ * E_ + e;
    float acc = 0.f;
    #pragma unroll 4
    for (int c = 0; c < C_; ++c) acc += wgt[c] * cp[(size_t)c * E_];
    out[(size_t)B_ * C_ * E_ + (size_t)b * E_ + e] = acc * inv;
  }
}

extern "C" void kernel_launch(void* const* d_in, const int* in_sizes, int n_in,
                              void* d_out, int out_size, void* d_ws, size_t ws_size,
                              hipStream_t stream) {
  const float* ctx = (const float*)d_in[0];   // (B, C, E) f32
  const float* qu  = (const float*)d_in[1];   // (B, Q, E) f32
  const float* ws  = (const float*)d_in[2];   // (3E,)     f32
  float* out = (float*)d_out;                 // c2q (B,C,E) then q2c (B,1,E)
  float* rowmax_ws = (float*)d_ws;            // B*C floats = 128 KB scratch

  dim3 g1(C_ / 128, B_);
  bidaf_sim_c2q_kernel<<<g1, dim3(256), 0, stream>>>(ctx, qu, ws, out, rowmax_ws);
  bidaf_q2c_kernel<<<dim3(B_), dim3(256), 0, stream>>>(ctx, rowmax_ws, out);
}
